// VFunction_52965536694891
// MI455X (gfx1250) — compile-verified, matrix operand order checked by reading, not run on hardware
//
#include <hip/hip_runtime.h>

// ---------------------------------------------------------------------------
// Fused value-function MLP (4x SiLU layers + scalar head) + GAE scan, CDNA5.
// GEMMs run on v_wmma_f32_16x16x32_f16 (f16 in, f32 accumulate).
// Weights converted/transposed once per launch into d_ws; activations stay
// in LDS per 32-token tile (no inter-layer HBM traffic).
// Register tile per wave: 32(M) x 64(N) = 8 WMMA accumulators per K-step,
// so each A fragment is reused by 4 B fragments (1.5 b128 loads per WMMA).
// ---------------------------------------------------------------------------

typedef _Float16 half_t;
typedef _Float16 v16h __attribute__((ext_vector_type(16)));
typedef _Float16 v8h  __attribute__((ext_vector_type(8)));
typedef float    v8f  __attribute__((ext_vector_type(8)));

#define T_DIM 64
#define B_DIM 512
#define D_DIM 1536
#define H_DIM 1024
#define N_TOK (T_DIM * B_DIM)   // 32768 tokens
#define ROWS  32                // tokens per workgroup
#define LDA   (D_DIM + 8)       // padded LDS stride (halves) for feat tile
#define LDH   (H_DIM + 8)       // padded LDS stride (halves) for hidden tiles

struct alignas(8) h4 { half_t v[4]; };

static __device__ __forceinline__ v16h cat8(v8h lo, v8h hi) {
    return __builtin_shufflevector(lo, hi, 0,1,2,3,4,5,6,7,8,9,10,11,12,13,14,15);
}

static __device__ __forceinline__ v8f wmma_f16(v16h a, v16h b, v8f c) {
    return __builtin_amdgcn_wmma_f32_16x16x32_f16(
        /*neg_a=*/false, a, /*neg_b=*/false, b,
        /*c_mod=*/(short)0, c, /*reuse_a=*/false, /*reuse_b=*/false);
}

// ---------------------------------------------------------------------------
// W (KxN, f32, row-major) -> Wt (NxK, f16) via LDS tile transpose.
// Both global read and write coalesced. Grid: (N/32, K/32), block: (32,8).
// ---------------------------------------------------------------------------
__global__ void w_transpose_f16(const float* __restrict__ W,
                                half_t* __restrict__ Wt, int K, int N) {
    __shared__ float tile[32][33];
    const int nb = blockIdx.x * 32;
    const int kb = blockIdx.y * 32;
    const int tx = threadIdx.x, ty = threadIdx.y;
#pragma unroll
    for (int j = 0; j < 32; j += 8) {
        tile[ty + j][tx] = W[(size_t)(kb + ty + j) * N + (nb + tx)];
    }
    __syncthreads();
#pragma unroll
    for (int j = 0; j < 32; j += 8) {
        Wt[(size_t)(nb + ty + j) * K + (kb + tx)] = (half_t)tile[tx][ty + j];
    }
}

// ---------------------------------------------------------------------------
// Fused MLP: 32 tokens per block, 8 waves; each wave owns 128 output columns
// processed as 2 passes of a 32x64 register tile.
// ---------------------------------------------------------------------------
__global__ __launch_bounds__(256)
void mlp_fused(const float* __restrict__ feat,
               const half_t* __restrict__ Wt0, const half_t* __restrict__ Wt1,
               const half_t* __restrict__ Wt2, const half_t* __restrict__ Wt3,
               const float* __restrict__ b0, const float* __restrict__ b1,
               const float* __restrict__ b2, const float* __restrict__ b3,
               const float* __restrict__ Wo, const float* __restrict__ bo,
               float* __restrict__ value) {
    extern __shared__ char smem[];
    half_t* bufA  = (half_t*)smem;                                   // 32 x LDA
    half_t* bufB  = (half_t*)(smem + (size_t)ROWS * LDA * 2);        // 32 x LDH
    float*  headP = (float*)(smem + (size_t)ROWS * LDA * 2 + (size_t)ROWS * LDH * 2);

    const int tid = threadIdx.x;
    const int n0  = blockIdx.x * ROWS;

    // ---- stage feat tile (f32 -> f16) into bufA -------------------------
    {
        const float4* f4 = (const float4*)(feat + (size_t)n0 * D_DIM);
        for (int v = tid; v < ROWS * (D_DIM / 4); v += 256) {
            const int r = v / (D_DIM / 4);
            const int c = v - r * (D_DIM / 4);
            float4 x = f4[(size_t)r * (D_DIM / 4) + c];
            h4 y;
            y.v[0] = (half_t)x.x; y.v[1] = (half_t)x.y;
            y.v[2] = (half_t)x.z; y.v[3] = (half_t)x.w;
            *(h4*)(bufA + (size_t)r * LDA + c * 4) = y;
        }
    }
    __syncthreads();

    const int wave   = tid >> 5;
    const int lane   = tid & 31;
    const int laneM  = lane & 15;     // row within 16 (M or N index)
    const int laneHi = lane >> 4;     // K-half selector

    const half_t* Wts[4]   = {Wt0, Wt1, Wt2, Wt3};
    const float*  bias_[4] = {b0, b1, b2, b3};

    half_t* in  = bufA; int ldIn = LDA;
    half_t* out = bufB;

#pragma unroll
    for (int l = 0; l < 4; ++l) {
        const int Kl = (l == 0) ? D_DIM : H_DIM;
        const half_t* Wt  = Wts[l];
        const float*  bia = bias_[l];

        for (int np = 0; np < 2; ++np) {
            const int nBase = wave * 128 + np * 64;

            v8f acc0[4] = {v8f{}, v8f{}, v8f{}, v8f{}};   // M rows 0..15
            v8f acc1[4] = {v8f{}, v8f{}, v8f{}, v8f{}};   // M rows 16..31

            const v8h* aRow0 = (const v8h*)(in + (size_t)laneM * ldIn + laneHi * 16);
            const v8h* aRow1 = (const v8h*)(in + (size_t)(16 + laneM) * ldIn + laneHi * 16);
            const v8h* bRow[4];
#pragma unroll
            for (int ni = 0; ni < 4; ++ni) {
                bRow[ni] = (const v8h*)(Wt + (size_t)(nBase + ni * 16 + laneM) * Kl
                                           + laneHi * 16);
            }

            for (int kk = 0; kk < Kl; kk += 32) {
                const int vi = kk >> 3;                  // v8h index for this K chunk
                v16h a0 = cat8(aRow0[vi], aRow0[vi + 1]);
                v16h a1 = cat8(aRow1[vi], aRow1[vi + 1]);
#pragma unroll
                for (int ni = 0; ni < 4; ++ni) {
                    v16h w = cat8(bRow[ni][vi], bRow[ni][vi + 1]);
                    acc0[ni] = wmma_f16(a0, w, acc0[ni]);
                    acc1[ni] = wmma_f16(a1, w, acc1[ni]);
                }
            }

            // epilogue: bias + SiLU -> f16 LDS (row-major, stride LDH)
#pragma unroll
            for (int ni = 0; ni < 4; ++ni) {
                const int   nc = nBase + ni * 16 + laneM;
                const float bv = bia[nc];
#pragma unroll
                for (int j = 0; j < 8; ++j) {
                    const int m = j + 8 * laneHi;
                    float x;
                    x = acc0[ni][j] + bv; x = x / (1.f + __expf(-x));
                    out[(size_t)m * LDH + nc] = (half_t)x;
                    x = acc1[ni][j] + bv; x = x / (1.f + __expf(-x));
                    out[(size_t)(16 + m) * LDH + nc] = (half_t)x;
                }
            }
        }
        __syncthreads();
        half_t* t = in; in = out; out = t;   // after l=0: in=bufB, out=bufA
        ldIn = LDH;
    }

    // ---- head: value[n] = dot(h3[n,:], Wo) + bo -------------------------
    {
        const int r   = tid >> 3;   // token row 0..31
        const int seg = tid & 7;    // 128-wide segment
        const v8h*   h8  = (const v8h*)(in + (size_t)r * LDH + seg * 128);
        const float* wsg = Wo + seg * 128;
        float s = 0.f;
#pragma unroll
        for (int c = 0; c < 16; ++c) {
            v8h hv = h8[c];
#pragma unroll
            for (int j = 0; j < 8; ++j) s += (float)hv[j] * wsg[c * 8 + j];
        }
        headP[r * 8 + seg] = s;
    }
    __syncthreads();
    if (tid < ROWS) {
        float s = bo[0];
#pragma unroll
        for (int j = 0; j < 8; ++j) s += headP[tid * 8 + j];
        value[n0 + tid] = s;
    }
}

// ---------------------------------------------------------------------------
// GAE reverse scan: one thread per batch column.
// ---------------------------------------------------------------------------
__global__ void gae_kernel(const float* __restrict__ value,
                           const float* __restrict__ reward,
                           const float* __restrict__ cont,
                           float* __restrict__ outp) {
    const int b = blockIdx.x * blockDim.x + threadIdx.x;
    if (b >= B_DIM) return;
    float* ret  = outp;
    float* base = outp + (T_DIM - 1) * B_DIM;
    float adv = 0.f;
    for (int t = T_DIM - 2; t >= 0; --t) {
        const float v    = value[t * B_DIM + b];
        const float vn   = value[(t + 1) * B_DIM + b];
        const float disc = cont[(t + 1) * B_DIM + b] * 0.997f;
        const float delta = reward[t * B_DIM + b] + disc * vn - v;
        adv = delta + disc * 0.95f * adv;
        ret[t * B_DIM + b]  = adv + v;
        base[t * B_DIM + b] = v;
    }
}

extern "C" void kernel_launch(void* const* d_in, const int* in_sizes, int n_in,
                              void* d_out, int out_size, void* d_ws, size_t ws_size,
                              hipStream_t stream) {
    (void)in_sizes; (void)n_in; (void)out_size; (void)ws_size;
    const float* feat   = (const float*)d_in[0];
    const float* reward = (const float*)d_in[1];
    const float* cont   = (const float*)d_in[2];
    const float* W0 = (const float*)d_in[3];
    const float* b0 = (const float*)d_in[4];
    const float* W1 = (const float*)d_in[5];
    const float* b1 = (const float*)d_in[6];
    const float* W2 = (const float*)d_in[7];
    const float* b2 = (const float*)d_in[8];
    const float* W3 = (const float*)d_in[9];
    const float* b3 = (const float*)d_in[10];
    const float* Wo = (const float*)d_in[11];
    const float* bo = (const float*)d_in[12];

    char* ws = (char*)d_ws;
    size_t off = 0;
    half_t* Wt0 = (half_t*)(ws + off); off += (size_t)H_DIM * D_DIM * sizeof(half_t);
    half_t* Wt1 = (half_t*)(ws + off); off += (size_t)H_DIM * H_DIM * sizeof(half_t);
    half_t* Wt2 = (half_t*)(ws + off); off += (size_t)H_DIM * H_DIM * sizeof(half_t);
    half_t* Wt3 = (half_t*)(ws + off); off += (size_t)H_DIM * H_DIM * sizeof(half_t);
    float*  value = (float*)(ws + off);

    const dim3 tb(32, 8);
    w_transpose_f16<<<dim3(H_DIM / 32, D_DIM / 32), tb, 0, stream>>>(W0, Wt0, D_DIM, H_DIM);
    w_transpose_f16<<<dim3(H_DIM / 32, H_DIM / 32), tb, 0, stream>>>(W1, Wt1, H_DIM, H_DIM);
    w_transpose_f16<<<dim3(H_DIM / 32, H_DIM / 32), tb, 0, stream>>>(W2, Wt2, H_DIM, H_DIM);
    w_transpose_f16<<<dim3(H_DIM / 32, H_DIM / 32), tb, 0, stream>>>(W3, Wt3, H_DIM, H_DIM);

    const size_t smem = (size_t)ROWS * LDA * 2 + (size_t)ROWS * LDH * 2 + (size_t)ROWS * 8 * 4;
    mlp_fused<<<N_TOK / ROWS, 256, smem, stream>>>(feat, Wt0, Wt1, Wt2, Wt3,
                                                   b0, b1, b2, b3, Wo, bo, value);

    gae_kernel<<<2, 256, 0, stream>>>(value, reward, cont, (float*)d_out);
}
